// GPT2SelfAttention_73400991089269
// MI455X (gfx1250) — matrix operand order repaired and checked by reading, not compiled, not run on hardware
//
#include <hip/hip_runtime.h>
#include <hip/hip_bf16.h>

// ---- GPT-2 self-attention: B=4, S=2048, D=1024, H=16, Dh=64
#define BB   4
#define SS   2048
#define DD   1024
#define HH   16
#define DH   64
#define D3   3072           // 3*D
#define BS   8192           // B*S
#define MASKV (-10000.0f)

typedef __attribute__((ext_vector_type(16))) __bf16 v16bf;
typedef __attribute__((ext_vector_type(8)))  float  v8f;
typedef __attribute__((ext_vector_type(4)))  unsigned int v4u;
typedef __attribute__((ext_vector_type(8)))  int    v8i;
typedef __attribute__((ext_vector_type(4)))  int    v4i;
typedef unsigned short u16;
typedef unsigned int   u32;

__device__ __forceinline__ u16 f2bf(float f) {
    __bf16 h = (__bf16)f;                      // native v_cvt path
    return __builtin_bit_cast(u16, h);
}

// ---- fragment loaders (ISA 7.12.2 VGPR layouts, wave32) --------------------

// A tile 16x32 (MxK), bf16 row-major, ldm = row stride (elements).
__device__ __forceinline__ v16bf load_tileA_bf16(const u16* __restrict__ base, int ldm, int lane) {
    int m = lane & 15, g = lane >> 4;
    const u16* p = base + (size_t)m * ldm + g * 8;
    union { v16bf v; uint4 q[2]; } r;
    r.q[0] = *reinterpret_cast<const uint4*>(p);
    r.q[1] = *reinterpret_cast<const uint4*>(p + 16);
    return r.v;
}

// A tile 16x32 from f32 source with inline cvt (softmax probs only).
__device__ __forceinline__ v16bf load_tileA_f32(const float* __restrict__ base, int ldm, int lane) {
    int m = lane & 15, g = lane >> 4;
    const float* p = base + (size_t)m * ldm + g * 8;
    union { v16bf v; __bf16 h[16]; } r;
#pragma unroll
    for (int j = 0; j < 8; ++j) {
        int k = (j < 4) ? (2 * j) : (16 + 2 * (j - 4));
        r.h[2 * j]     = (__bf16)p[k];
        r.h[2 * j + 1] = (__bf16)p[k + 1];
    }
    return r.v;
}

// B tile 32x16 (KxN) where memory is [N][K] row-major bf16, ldk = row stride.
__device__ __forceinline__ v16bf load_tileBt_bf16(const u16* __restrict__ base, int ldk, int lane) {
    int n = lane & 15, g = lane >> 4;
    const u16* p = base + (size_t)n * ldk + g * 8;
    union { v16bf v; uint4 q[2]; } r;
    r.q[0] = *reinterpret_cast<const uint4*>(p);
    r.q[1] = *reinterpret_cast<const uint4*>(p + 16);
    return r.v;
}

__device__ __forceinline__ v8f wmma_bf16(v16bf a, v16bf b, v8f c) {
    return __builtin_amdgcn_wmma_f32_16x16x32_bf16(false, a, false, b, (short)0, c, false, false);
}

// ---- TDM: load one 16x32-bf16 tile (rows stride K elems) into LDS ----------
// D# per CDNA5 ISA ch.8: group0 = {count, lds_addr, global_addr, type=2};
// group1 = {data_size=2B, tensor_dim0=32, tensor_dim1=16, tile_dim0=32,
//           tile_dim1=16, tensor_dim0_stride=K}. Groups 2/3/extra unused (2-D).
__device__ __forceinline__ void tdm_load_tile(u32 ldsAddr, const u16* gptr, int K) {
    unsigned long long ga = (unsigned long long)(size_t)gptr;
    v4u g0;
    g0.x = 1u;                                             // count=1 (valid), user mode
    g0.y = ldsAddr;                                        // LDS byte address
    g0.z = (u32)ga;                                        // global_addr[31:0]
    g0.w = (u32)((ga >> 32) & 0x01FFFFFFu) | (2u << 30);   // addr[56:32] | type=2
    v8i g1;
    g1[0] = (1 << 16);                                     // data_size = 1 -> 2 bytes
    g1[1] = (32 << 16);                                    // tensor_dim0 = 32 (bits 79:48)
    g1[2] = (16 << 16);                                    // tensor_dim1 = 16 (bits 111:80)
    g1[3] = (32 << 16);                                    // tile_dim0 = 32 (bits 127:112)
    g1[4] = 16;                                            // tile_dim1 = 16 (bits 143:128)
    g1[5] = K;                                             // tensor_dim0_stride[31:0] (elems)
    g1[6] = 0;
    g1[7] = 0;
    // 6-arg form (clang-23 / therock headers): groups 2, 3, extra group, cpol
    __builtin_amdgcn_tensor_load_to_lds(g0, g1, (v4i)0, (v4i)0, (v8i)0, 0);
}

// ---- conversion / transpose helpers ---------------------------------------

__global__ void cvt_bf16(const float* __restrict__ src, u16* __restrict__ dst, int n) {
    int i = blockIdx.x * 256 + threadIdx.x;
    if (i < n) dst[i] = f2bf(src[i]);
}

// src [K x N] f32 -> dst [N x K] bf16. grid = (K/256, N).
__global__ void transpose_cvt(const float* __restrict__ src, u16* __restrict__ dst, int K, int N) {
    int k = blockIdx.x * 256 + threadIdx.x;
    int n = blockIdx.y;
    dst[(size_t)n * K + k] = f2bf(src[(size_t)k * N + n]);
}

// qkv_bf16 [B*S x 3D] -> Vt [B,H,Dh,S] bf16. grid = (S/256, B*H*Dh).
__global__ void transpose_v(const u16* __restrict__ qkv, u16* __restrict__ Vt) {
    int s   = blockIdx.x * 256 + threadIdx.x;
    int row = blockIdx.y;                 // bh*64 + d
    int bh = row >> 6, d = row & 63;
    int b = bh >> 4, h = bh & 15;
    Vt[(size_t)row * SS + s] = qkv[((size_t)b * SS + s) * D3 + 2 * DD + h * DH + d];
}

// ---------------------------------------------------------------------------
// Dense GEMM: C[M x N] = A[M x K](bf16) * Wt[N x K](bf16)ᵀ + bias.
// 8 waves/block share the N-tile. Wave 0 drives a double-buffered TDM pipeline
// (tensor_load_to_lds + s_wait_tensorcnt) staging the 1 KB B tile in LDS;
// every wave reads its B fragment back as two ds b128 loads.
// grid = (N/16, M/128), block = 256.
template <bool STORE_BF16>
__global__ void gemm_bias_t(const u16* __restrict__ A, const u16* __restrict__ Wt,
                            const float* __restrict__ bias, void* __restrict__ Cout,
                            int lda, int K, int ldc) {
    __shared__ __align__(16) u32 lds[2][256];
    int tid  = threadIdx.x;
    int lane = tid & 31, wave = tid >> 5;
    int mBase = (blockIdx.y * 8 + wave) * 16;
    int nBase = blockIdx.x * 16;
    bool issuer = (wave == 0);

    u32 ldsA0 = (u32)(size_t)&lds[0][0];
    u32 ldsA1 = (u32)(size_t)&lds[1][0];
    const u16* wTile = Wt + (size_t)nBase * K;   // tile origin at [nBase][0]

    float bv = bias[nBase + (lane & 15)];
    v8f acc;
#pragma unroll
    for (int r = 0; r < 8; ++r) acc[r] = bv;

    const u16* pA = A + (size_t)mBase * lda;
    int nsteps = K / 32;

    if (issuer) {
        tdm_load_tile(ldsA0, wTile, K);
        __builtin_amdgcn_s_wait_tensorcnt(0);
    }
    __syncthreads();

    for (int s = 0; s < nsteps; ++s) {
        int buf = s & 1;
        if (issuer && (s + 1 < nsteps))
            tdm_load_tile(buf ? ldsA0 : ldsA1, wTile + (s + 1) * 32, K);

        __builtin_prefetch(pA + (s + 2) * 32, 0, 0);
        v16bf a = load_tileA_bf16(pA + s * 32, lda, lane);

        union { v16bf v; uint4 q[2]; } bb;
        const uint4* sp = reinterpret_cast<const uint4*>(
            &lds[buf][(lane & 15) * 16 + (lane >> 4) * 4]);
        bb.q[0] = sp[0];
        bb.q[1] = sp[2];

        acc = wmma_bf16(a, bb.v, acc);

        if (issuer) __builtin_amdgcn_s_wait_tensorcnt(0);
        __syncthreads();
    }

    int row0 = mBase + 8 * (lane >> 4);
    int col  = nBase + (lane & 15);
    if (STORE_BF16) {
        u16* C = (u16*)Cout;
#pragma unroll
        for (int r = 0; r < 8; ++r)
            C[(size_t)(row0 + r) * ldc + col] = f2bf(acc[r]);
    } else {
        float* C = (float*)Cout;
#pragma unroll
        for (int r = 0; r < 8; ++r)
            C[(size_t)(row0 + r) * ldc + col] = acc[r];
    }
}

// ---------------------------------------------------------------------------
// Scores: attnW[bh][q][k] = Q·Kᵀ/8, causal -> MASKV, + attn_mask[b][k] (f32 out).
__global__ void attn_scores(const u16* __restrict__ qkv, const float* __restrict__ mask,
                            float* __restrict__ attnW) {
    int lane = threadIdx.x & 31;
    int wave = threadIdx.x >> 5;
    int bh = blockIdx.z;
    int b = bh >> 4, h = bh & 15;
    int qTile = blockIdx.y * 8 + wave;
    int kTile = blockIdx.x;
    int qBase = qTile * 16, kBase = kTile * 16;
    int g = lane >> 4, n = lane & 15;

    float* dst = attnW + ((size_t)bh * SS + qBase) * SS + kBase;
    const float* maskRow = mask + (size_t)b * SS;

    if (kTile > qTile) {                  // fully masked tile: store only
        float mv = MASKV + maskRow[kBase + n];
#pragma unroll
        for (int r = 0; r < 8; ++r)
            dst[(size_t)(8 * g + r) * SS + n] = mv;
        return;
    }

    const u16* Q  = qkv + ((size_t)b * SS + qBase) * D3 + h * DH;
    const u16* Kp = qkv + ((size_t)b * SS + kBase) * D3 + DD + h * DH;

    v8f acc = {};
#pragma unroll
    for (int d0 = 0; d0 < DH; d0 += 32) {
        v16bf a  = load_tileA_bf16(Q + d0, D3, lane);
        v16bf bt = load_tileBt_bf16(Kp + d0, D3, lane);   // B[d][n] = K[n][d]
        acc = wmma_bf16(a, bt, acc);
    }

    int kk = kBase + n;
    float madd = maskRow[kk];
#pragma unroll
    for (int r = 0; r < 8; ++r) {
        int qq = qBase + 8 * g + r;
        float s = acc[r] * 0.125f;                        // 1/sqrt(64)
        if (kk > qq) s = MASKV;
        dst[(size_t)(8 * g + r) * SS + n] = s + madd;
    }
}

// ---------------------------------------------------------------------------
// In-place row softmax over 2048 cols; one 256-thread block per row.
__global__ void softmax_rows(float* __restrict__ attnW) {
    float* p = attnW + (size_t)blockIdx.x * SS;
    __shared__ float red[8];
    int tid = threadIdx.x;

    float v[8];
    float m = -3.402823466e38f;
#pragma unroll
    for (int i = 0; i < 8; ++i) { v[i] = p[tid + 256 * i]; m = fmaxf(m, v[i]); }
#pragma unroll
    for (int off = 16; off; off >>= 1) m = fmaxf(m, __shfl_xor(m, off, 32));
    if ((tid & 31) == 0) red[tid >> 5] = m;
    __syncthreads();
    m = red[0];
#pragma unroll
    for (int w = 1; w < 8; ++w) m = fmaxf(m, red[w]);

    float s = 0.f;
#pragma unroll
    for (int i = 0; i < 8; ++i) { v[i] = __expf(v[i] - m); s += v[i]; }
#pragma unroll
    for (int off = 16; off; off >>= 1) s += __shfl_xor(s, off, 32);
    __syncthreads();
    if ((tid & 31) == 0) red[tid >> 5] = s;
    __syncthreads();
    s = 0.f;
#pragma unroll
    for (int w = 0; w < 8; ++w) s += red[w];

    float inv = 1.0f / s;
#pragma unroll
    for (int i = 0; i < 8; ++i) p[tid + 256 * i] = v[i] * inv;
}

// ---------------------------------------------------------------------------
// PV: heads[b][q][h][d] = sum_k P[bh][q][k] * V[b][k][h*64+d]  (bf16 out).
__global__ void attn_pv(const float* __restrict__ attnW, const u16* __restrict__ Vt,
                        u16* __restrict__ outH) {
    int lane = threadIdx.x & 31;
    int wave = threadIdx.x >> 5;
    int bh = blockIdx.y;
    int b = bh >> 4, h = bh & 15;
    int qTile = blockIdx.x * 8 + wave;
    int qBase = qTile * 16;

    const float* P = attnW + ((size_t)bh * SS + qBase) * SS;
    const u16*  V0 = Vt + (size_t)bh * DH * SS;

    v8f a0 = {}, a1 = {}, a2 = {}, a3 = {};
    int kEnd = qBase + 16;                          // causal: P==0 for k>q
    for (int k0 = 0; k0 < kEnd; k0 += 32) {
        __builtin_prefetch(P + k0 + 64, 0, 0);
        v16bf a  = load_tileA_f32(P + k0, SS, lane);
        v16bf b0 = load_tileBt_bf16(V0 + (size_t)(0 * 16) * SS + k0, SS, lane);
        v16bf b1 = load_tileBt_bf16(V0 + (size_t)(1 * 16) * SS + k0, SS, lane);
        v16bf b2 = load_tileBt_bf16(V0 + (size_t)(2 * 16) * SS + k0, SS, lane);
        v16bf b3 = load_tileBt_bf16(V0 + (size_t)(3 * 16) * SS + k0, SS, lane);
        a0 = wmma_bf16(a, b0, a0);
        a1 = wmma_bf16(a, b1, a1);
        a2 = wmma_bf16(a, b2, a2);
        a3 = wmma_bf16(a, b3, a3);
    }

    int g = lane >> 4, n = lane & 15;
#pragma unroll
    for (int r = 0; r < 8; ++r) {
        size_t q = qBase + 8 * g + r;
        u16* o = outH + (((size_t)b * SS + q) * HH + h) * DH + n;
        o[0]  = f2bf(a0[r]);
        o[16] = f2bf(a1[r]);
        o[32] = f2bf(a2[r]);
        o[48] = f2bf(a3[r]);
    }
}

// ---------------------------------------------------------------------------
extern "C" void kernel_launch(void* const* d_in, const int* in_sizes, int n_in,
                              void* d_out, int out_size, void* d_ws, size_t ws_size,
                              hipStream_t stream) {
    const float* x         = (const float*)d_in[0];
    const float* attn_mask = (const float*)d_in[1];
    const float* W_in      = (const float*)d_in[2];
    const float* b_in      = (const float*)d_in[3];
    const float* W_out     = (const float*)d_in[4];
    const float* b_out     = (const float*)d_in[5];

    float* out   = (float*)d_out;                    // [B,S,D] f32
    float* attnW = out + (size_t)BS * DD;            // [B,H,S,S] f32

    // workspace carve-up (bf16 buffers), 104 MB total
    u16* xb    = (u16*)d_ws;                         // [B*S, D]      16 MB
    u16* WinT  = xb    + (size_t)BS * DD;            // [3D, D]        6 MB
    u16* WoutT = WinT  + (size_t)D3 * DD;            // [D, D]         2 MB
    u16* qkv   = WoutT + (size_t)DD * DD;            // [B*S, 3D]     48 MB
    u16* Vt    = qkv   + (size_t)BS * D3;            // [B,H,Dh,S]    16 MB
    u16* outH  = Vt    + (size_t)BB * HH * DH * SS;  // [B,S,H,Dh]    16 MB

    dim3 blk(256);

    // 0) one-time conversions / transposes
    cvt_bf16<<<dim3((BS * DD) / 256), blk, 0, stream>>>(x, xb, BS * DD);
    transpose_cvt<<<dim3(DD / 256, D3), blk, 0, stream>>>(W_in,  WinT,  DD, D3);
    transpose_cvt<<<dim3(DD / 256, DD), blk, 0, stream>>>(W_out, WoutT, DD, DD);

    // 1) qkv = x @ W_in + b_in   (bf16 out)    M=8192 N=3072 K=1024
    gemm_bias_t<true><<<dim3(D3 / 16, BS / 128), blk, 0, stream>>>(
        xb, WinT, b_in, qkv, DD, DD, D3);

    // 1b) Vt = transpose(V slice of qkv)
    transpose_v<<<dim3(SS / 256, BB * HH * DH), blk, 0, stream>>>(qkv, Vt);

    // 2) raw scores + causal + mask -> attnW (f32, in d_out)
    attn_scores<<<dim3(SS / 16, SS / 128, BB * HH), blk, 0, stream>>>(
        qkv, attn_mask, attnW);

    // 3) softmax in place
    softmax_rows<<<dim3(BB * HH * SS), blk, 0, stream>>>(attnW);

    // 4) heads = P @ V (merged-head bf16 layout)
    attn_pv<<<dim3(SS / 128, BB * HH), blk, 0, stream>>>(attnW, Vt, outH);

    // 5) out = heads @ W_out + b_out (f32)     M=8192 N=1024 K=1024
    gemm_bias_t<false><<<dim3(DD / 16, BS / 128), blk, 0, stream>>>(
        outH, WoutT, b_out, out, DD, DD, DD);
}
